// ASSANetDecoder_6193342841556
// MI455X (gfx1250) — matrix operand order repaired and checked by reading, not compiled, not run on hardware
//
#include <hip/hip_runtime.h>
#include <hip/hip_bf16.h>
#include <stddef.h>

// ---------------------------------------------------------------------------
// ASSANet decoder for MI455X (gfx1250, wave32).
// All 1x1-conv layers run as V_WMMA_F32_16X16X32_BF16 GEMMs (fp32 accumulate)
// with fused BN+ReLU epilogues. Each wave computes a 16x64 output tile
// (4 accumulators sharing one A fragment) for 4x A-operand reuse.
// Skip features are transposed through LDS for fully coalesced traffic.
// ---------------------------------------------------------------------------

typedef __attribute__((ext_vector_type(16))) __bf16 v16bf;
typedef __attribute__((ext_vector_type(8)))  __bf16 v8bf;
typedef __attribute__((ext_vector_type(8)))  float  v8f;

union FragU { v16bf v; v8bf h[2]; };

#define BATCH 8
#define WPB 4            // waves per block in the GEMM

// ---------------------------- prep kernels ---------------------------------

__global__ void cvt_bf16_kernel(const float* __restrict__ src,
                                __bf16* __restrict__ dst, int nelem) {
  int i = blockIdx.x * blockDim.x + threadIdx.x;
  if (i < nelem) dst[i] = (__bf16)src[i];
}

__global__ void bn_prep_kernel(const float* __restrict__ g,
                               const float* __restrict__ be,
                               const float* __restrict__ mu,
                               const float* __restrict__ va,
                               float* __restrict__ sc,
                               float* __restrict__ sh, int c) {
  int i = blockIdx.x * blockDim.x + threadIdx.x;
  if (i < c) {
    float s = g[i] / sqrtf(va[i] + 1e-5f);
    sc[i] = s;
    sh[i] = be[i] - mu[i] * s;
  }
}

// ------------------------------- 3-NN --------------------------------------

#define KNN_TILE 256

__global__ void knn3_kernel(const float* __restrict__ uxyz,  // [B][n][3]
                            const float* __restrict__ kxyz,  // [B][m][3]
                            int n, int m,
                            int*   __restrict__ idx_out,     // [B][n][3]
                            float* __restrict__ w_out) {     // [B][n][3]
  __shared__ float sx[KNN_TILE], sy[KNN_TILE], sz[KNN_TILE], sn[KNN_TILE];
  const int b = blockIdx.y;
  const int i = blockIdx.x * blockDim.x + threadIdx.x;
  const bool valid = (i < n);
  float ux = 0.f, uy = 0.f, uz = 0.f, un = 0.f;
  if (valid) {
    const float* u = uxyz + ((size_t)b * n + i) * 3;
    ux = u[0]; uy = u[1]; uz = u[2];
    un = ux * ux + uy * uy + uz * uz;   // same decomposition as reference
  }
  float b0 = 3.4e38f, b1 = 3.4e38f, b2 = 3.4e38f;
  int   j0 = 0, j1 = 0, j2 = 0;
  for (int t0 = 0; t0 < m; t0 += KNN_TILE) {
    const int cnt = min(KNN_TILE, m - t0);
    __syncthreads();
    for (int t = threadIdx.x; t < cnt; t += blockDim.x) {
      const float* k = kxyz + ((size_t)b * m + t0 + t) * 3;
      float kx = k[0], ky = k[1], kz = k[2];
      sx[t] = kx; sy[t] = ky; sz[t] = kz;
      sn[t] = kx * kx + ky * ky + kz * kz;
    }
    __syncthreads();
    if (valid) {
      for (int t = 0; t < cnt; ++t) {
        float d = un + sn[t] - 2.0f * (ux * sx[t] + uy * sy[t] + uz * sz[t]);
        int j = t0 + t;
        if (d < b0)      { b2 = b1; j2 = j1; b1 = b0; j1 = j0; b0 = d; j0 = j; }
        else if (d < b1) { b2 = b1; j2 = j1; b1 = d;  j1 = j; }
        else if (d < b2) { b2 = d;  j2 = j; }
      }
    }
  }
  if (valid) {
    float r0 = 1.0f / (sqrtf(fmaxf(b0, 0.f)) + 1e-8f);
    float r1 = 1.0f / (sqrtf(fmaxf(b1, 0.f)) + 1e-8f);
    float r2 = 1.0f / (sqrtf(fmaxf(b2, 0.f)) + 1e-8f);
    float s = 1.0f / (r0 + r1 + r2);
    size_t o = ((size_t)b * n + i) * 3;
    idx_out[o] = j0; idx_out[o + 1] = j1; idx_out[o + 2] = j2;
    w_out[o] = r0 * s; w_out[o + 1] = r1 * s; w_out[o + 2] = r2 * s;
  }
}

// -------------------- interpolation -> bf16 (first C2 channels) ------------

__global__ void interp_kernel(const float* __restrict__ kn,
                              size_t sb, int sc, int sj,
                              const int*   __restrict__ idx,
                              const float* __restrict__ w,
                              __bf16* __restrict__ Xc,        // [B][n][C2+C1]
                              int n, int C1, int C2) {
  const int b = blockIdx.y;
  const int i = blockIdx.x;
  size_t o = ((size_t)b * n + i) * 3;
  const int   j0 = idx[o], j1 = idx[o + 1], j2 = idx[o + 2];
  const float w0 = w[o], w1 = w[o + 1], w2 = w[o + 2];
  const float* kb = kn + (size_t)b * sb;
  __bf16* xrow = Xc + ((size_t)b * n + i) * (size_t)(C1 + C2);
  for (int c = threadIdx.x; c < C2; c += blockDim.x) {
    float v = w0 * kb[(size_t)c * sc + (size_t)j0 * sj]
            + w1 * kb[(size_t)c * sc + (size_t)j1 * sj]
            + w2 * kb[(size_t)c * sc + (size_t)j2 * sj];
    xrow[c] = (__bf16)v;
  }
}

// ---------------- skip features: coalesced LDS transpose -------------------
// Xc[b][i][C2+c] = skip[b][c][i], 32x32 float tiles.

__global__ void skip_transpose_kernel(const float* __restrict__ skip, // [B][C1][n]
                                      __bf16* __restrict__ Xc,        // [B][n][Cin]
                                      int n, int C1, int C2) {
  __shared__ float tile[32][33];
  const int b  = blockIdx.z;
  const int i0 = blockIdx.x << 5;
  const int c0 = blockIdx.y << 5;
  const int tx = threadIdx.x;     // 0..31
  const int ty = threadIdx.y;     // 0..7
  const int Cin = C1 + C2;
#pragma unroll
  for (int r = 0; r < 4; ++r) {
    int c = c0 + ty + r * 8;
    tile[ty + r * 8][tx] = skip[((size_t)b * C1 + c) * n + i0 + tx];
  }
  __syncthreads();
#pragma unroll
  for (int r = 0; r < 4; ++r) {
    int i = i0 + ty + r * 8;
    Xc[((size_t)b * n + i) * Cin + C2 + c0 + tx] = (__bf16)tile[tx][ty + r * 8];
  }
}

// ------------------- WMMA GEMM + BN + ReLU epilogue ------------------------
// D = relu(bn(W @ X)).  W: [M][K] bf16 row-major.  X: [B][n][K] bf16.
// One wave computes a 16(M) x 64(N) tile: 4 accumulators share one A frag.
// out_mode: 0 = bf16 [B][n][M], 1 = f32 [B][n][M], 2 = f32 [B][M][n].

__global__ void gemm_bn_relu_kernel(const __bf16* __restrict__ Wb,
                                    const __bf16* __restrict__ X,
                                    const float* __restrict__ scale,
                                    const float* __restrict__ shift,
                                    void* __restrict__ out,
                                    int M, int K, int n, int out_mode) {
  const int lane = threadIdx.x & 31;
  const int wave = threadIdx.x >> 5;
  const int half = lane >> 4;
  const int l    = lane & 15;
  const int ngroups = n >> 6;                 // 64-column groups
  const int group = blockIdx.x * WPB + wave;  // wave-uniform
  if (group >= ngroups) return;
  const int m0 = blockIdx.y << 4;
  const int b  = blockIdx.z;
  const int colbase = (group << 6) + l;

  // A fragment (16x32 bf16): lanes 0-15 row=l hold K {kk..kk+7, kk+16..kk+23},
  // lanes 16-31 row=l hold K {kk+8..kk+15, kk+24..kk+31}.
  const __bf16* wrow = Wb + (size_t)(m0 + l) * K + half * 8;
  // B fragment (32x16 bf16): lanes 0-15 col N=l hold K {kk..kk+7, kk+16..kk+23},
  // lanes 16-31 hold K {kk+8..kk+15, kk+24..kk+31}.  4 column tiles, stride 16*K elements.
  const __bf16* xcol = X + ((size_t)b * n + colbase) * K + half * 8;
  const size_t xs = (size_t)16 * K;

  v8f acc[4] = {{}, {}, {}, {}};
  for (int kk = 0; kk < K; kk += 32) {
    FragU a;
    a.h[0] = *(const v8bf*)(wrow + kk);
    a.h[1] = *(const v8bf*)(wrow + kk + 16);
    FragU bf[4];
#pragma unroll
    for (int t = 0; t < 4; ++t) {
      bf[t].h[0] = *(const v8bf*)(xcol + (size_t)t * xs + kk);
      bf[t].h[1] = *(const v8bf*)(xcol + (size_t)t * xs + kk + 16);
    }
    if (kk + 128 < K) {  // stream the 4 B columns ahead (global_prefetch_b8)
#pragma unroll
      for (int t = 0; t < 4; ++t)
        __builtin_prefetch((const void*)(xcol + (size_t)t * xs + kk + 128), 0, 3);
    }
#pragma unroll
    for (int t = 0; t < 4; ++t)
      acc[t] = __builtin_amdgcn_wmma_f32_16x16x32_bf16(
          /*neg_a=*/false, a.v, /*neg_b=*/false, bf[t].v,
          /*c_mod=*/(short)0, acc[t], /*reuse_a=*/false, /*reuse_b=*/false);
  }

  // BN params for this lane's 8 rows (rows m0 + 8*half + r).
  float sc8[8], sh8[8];
#pragma unroll
  for (int r = 0; r < 8; ++r) {
    const int mrow = m0 + half * 8 + r;
    sc8[r] = scale[mrow];
    sh8[r] = shift[mrow];
  }

  // C/D layout: VGPR r -> row m0 + 8*half + r, col = tile base + l.
#pragma unroll
  for (int t = 0; t < 4; ++t) {
    const int col = colbase + t * 16;
#pragma unroll
    for (int r = 0; r < 8; ++r) {
      const int mrow = m0 + half * 8 + r;
      float v = acc[t][r] * sc8[r] + sh8[r];
      v = v > 0.f ? v : 0.f;
      if (out_mode == 0)
        ((__bf16*)out)[((size_t)b * n + col) * M + mrow] = (__bf16)v;
      else if (out_mode == 1)
        ((float*)out)[((size_t)b * n + col) * M + mrow] = v;
      else
        ((float*)out)[((size_t)b * M + mrow) * n + col] = v;
    }
  }
}

// ------------------------------- host --------------------------------------

extern "C" void kernel_launch(void* const* d_in, const int* in_sizes, int n_in,
                              void* d_out, int out_size, void* d_ws, size_t ws_size,
                              hipStream_t stream) {
  static const int NSv[5] = {8192, 2048, 512, 128, 32};
  static const int CHv[5] = {64, 128, 256, 512, 1024};
  static const int MLP[4][3] = {{320, 128, 128}, {640, 256, 256},
                                {768, 512, 512}, {1536, 512, 512}};

  const float* xyz[5];
  const float* feat[5];
  for (int l = 0; l < 5; ++l) {
    xyz[l]  = (const float*)d_in[l];
    feat[l] = (const float*)d_in[5 + l];
  }
  // params flattened: for mlpIdx 0..3, layer 0..1: (W, gamma, beta, mu, var)
  const float *Wf[4][2], *gam[4][2], *bet[4][2], *muu[4][2], *var[4][2];
  int pi = 10;
  for (int mI = 0; mI < 4; ++mI)
    for (int L = 0; L < 2; ++L) {
      Wf[mI][L]  = (const float*)d_in[pi++];
      gam[mI][L] = (const float*)d_in[pi++];
      bet[mI][L] = (const float*)d_in[pi++];
      muu[mI][L] = (const float*)d_in[pi++];
      var[mI][L] = (const float*)d_in[pi++];
    }

  // workspace carve-up
  char* p = (char*)d_ws;
  auto take = [&](size_t bytes) -> void* {
    void* r = (void*)p;
    p += (bytes + 255) & ~(size_t)255;
    return r;
  };
  __bf16* wb[4][2];
  float*  bsc[4][2];
  float*  bsh[4][2];
  for (int mI = 0; mI < 4; ++mI)
    for (int L = 0; L < 2; ++L) {
      int cout = MLP[mI][L + 1], cin = MLP[mI][L];
      wb[mI][L]  = (__bf16*)take((size_t)cout * cin * sizeof(__bf16));
      bsc[mI][L] = (float*)take((size_t)cout * sizeof(float));
      bsh[mI][L] = (float*)take((size_t)cout * sizeof(float));
    }
  __bf16* Xcat = (__bf16*)take((size_t)BATCH * 8192 * 320 * sizeof(__bf16));
  __bf16* Hbuf = (__bf16*)take((size_t)BATCH * 8192 * 128 * sizeof(__bf16));
  float*  O3   = (float*)take((size_t)BATCH * 128  * 512 * sizeof(float));
  float*  O2   = (float*)take((size_t)BATCH * 512  * 512 * sizeof(float));
  float*  O1   = (float*)take((size_t)BATCH * 2048 * 256 * sizeof(float));
  int*    kidx = (int*)  take((size_t)BATCH * 8192 * 3 * sizeof(int));
  float*  kw   = (float*)take((size_t)BATCH * 8192 * 3 * sizeof(float));

  // weight conversion + BN folding
  for (int mI = 0; mI < 4; ++mI)
    for (int L = 0; L < 2; ++L) {
      int cout = MLP[mI][L + 1], cin = MLP[mI][L];
      int nel = cout * cin;
      cvt_bf16_kernel<<<(nel + 255) / 256, 256, 0, stream>>>(Wf[mI][L], wb[mI][L], nel);
      bn_prep_kernel<<<(cout + 255) / 256, 256, 0, stream>>>(
          gam[mI][L], bet[mI][L], muu[mI][L], var[mI][L], bsc[mI][L], bsh[mI][L], cout);
    }

  // stage table: (unknown level, known level, mlp index), coarse -> fine
  const int stLU[4] = {3, 2, 1, 0};
  const int stLK[4] = {4, 3, 2, 1};
  const int stMI[4] = {3, 2, 1, 0};
  float* Obuf[3] = {O3, O2, O1};

  const float* kn_feat = feat[4];   // stage A knowns: original (B,C2,m) layout
  int C2 = CHv[4];

  for (int s = 0; s < 4; ++s) {
    const int lu = stLU[s], lk = stLK[s], mI = stMI[s];
    const int n = NSv[lu], m = NSv[lk];
    const int C1 = CHv[lu];
    const int Cin = MLP[mI][0];      // == C1 + C2
    const int Chid = MLP[mI][1];
    const int Cout = MLP[mI][2];

    // 3-NN
    dim3 kg((n + 255) / 256, BATCH);
    knn3_kernel<<<kg, 256, 0, stream>>>(xyz[lu], xyz[lk], n, m, kidx, kw);

    // interpolation -> first C2 channels of Xcat
    size_t sb; int sc, sj;
    if (s == 0) { sb = (size_t)C2 * m; sc = m; sj = 1; }   // channel-major input
    else        { sb = (size_t)m * C2; sc = 1; sj = C2; }  // point-major intermediate
    interp_kernel<<<dim3(n, BATCH), 128, 0, stream>>>(
        kn_feat, sb, sc, sj, kidx, kw, Xcat, n, C1, C2);

    // skip features -> last C1 channels of Xcat (coalesced LDS transpose)
    skip_transpose_kernel<<<dim3(n / 32, C1 / 32, BATCH), dim3(32, 8), 0, stream>>>(
        feat[lu], Xcat, n, C1, C2);

    // layer 1: hidden = relu(bn(W1 @ Xcat)) -> bf16 point-major
    dim3 g1((n / 64 + WPB - 1) / WPB, Chid / 16, BATCH);
    gemm_bn_relu_kernel<<<g1, WPB * 32, 0, stream>>>(
        wb[mI][0], Xcat, bsc[mI][0], bsh[mI][0], (void*)Hbuf, Chid, Cin, n, 0);

    // layer 2: out = relu(bn(W2 @ hidden))
    const int mode = (s == 3) ? 2 : 1;               // final: (B,M,n) fp32
    void* outp = (s == 3) ? d_out : (void*)Obuf[s];
    dim3 g2((n / 64 + WPB - 1) / WPB, Cout / 16, BATCH);
    gemm_bn_relu_kernel<<<g2, WPB * 32, 0, stream>>>(
        wb[mI][1], Hbuf, bsc[mI][1], bsh[mI][1], outp, Cout, Chid, n, mode);

    kn_feat = Obuf[s];   // next stage's knowns (fp32 point-major)
    C2 = Cout;
  }
}
